// MultiHA_52106543235815
// MI455X (gfx1250) — matrix operand order repaired and checked by reading, not compiled, not run on hardware
//
#include <hip/hip_runtime.h>
#include <hip/hip_fp16.h>

typedef __attribute__((ext_vector_type(16))) _Float16 v16h;
typedef __attribute__((ext_vector_type(8)))  _Float16 v8h;
typedef __attribute__((ext_vector_type(8)))  float    v8f;

#define KD  1024
#define KS  2048
#define KB  4
#define KNH 16
#define KHS 64
#define NEG_BIG (-1.0e30f)

// ---- WMMA fragment loaders (contiguous k order, matched by the B-fragment loads) ----
// A 16x32 f16: lane m=L&15, g=L>>4; halves 0..15 -> K=c0+16g.. (contiguous)
static __device__ __forceinline__ v16h a_frag_from_f32(const float* __restrict__ arow, int c0, int ag) {
  v16h a;
  const float* p0 = arow + c0 + 16 * ag;
  const float* p1 = arow + c0 + 16 * ag + 8;
#pragma unroll
  for (int i = 0; i < 8; ++i) { a[i] = (_Float16)p0[i]; a[8 + i] = (_Float16)p1[i]; }
  return a;
}
static __device__ __forceinline__ v16h a_frag_from_f16(const _Float16* __restrict__ arow, int c0, int ag) {
  v16h a;
  v8h lo = *(const v8h*)(arow + c0 + 16 * ag);
  v8h hi = *(const v8h*)(arow + c0 + 16 * ag + 8);
#pragma unroll
  for (int i = 0; i < 8; ++i) { a[i] = lo[i]; a[8 + i] = hi[i]; }
  return a;
}

// =====================================================================
// fp32 -> fp16 weight conversion (one-time; keeps GEMM B-frags cvt-free)
// =====================================================================
__global__ __launch_bounds__(256)
void cvt_kernel(const float* __restrict__ src, _Float16* __restrict__ dst, int n) {
  int i = (blockIdx.x * 256 + threadIdx.x) * 8;
  if (i + 8 <= n) {
    v8f v = *(const v8f*)(src + i);
    v8h o;
#pragma unroll
    for (int j = 0; j < 8; ++j) o[j] = (_Float16)v[j];
    *(v8h*)(dst + i) = o;
  }
}

// =====================================================================
// Projection: Out = A(f32)[N,KD] @ W16[Dout,KD]^T  ->  f16
//   mode 0: dst[((b*NH+h)*S + s)*HS + e]   (Q,K head-major)
//   mode 1: dst[((b*NH+h)*HS + e)*S + s]   (V transposed per head)
// grid (N/32, 4), block 128 (4 waves); wave = 32x64 output tile, 8 wmma/chunk
// =====================================================================
__global__ __launch_bounds__(128)
void proj_kernel(const float* __restrict__ A, const _Float16* __restrict__ W16,
                 _Float16* __restrict__ dst, int mode) {
  const int lane = threadIdx.x & 31;
  const int wid  = threadIdx.x >> 5;
  const int row0 = blockIdx.x * 32;
  const int col0 = (blockIdx.y * 4 + wid) * 64;
  const int am = lane & 15, ag = lane >> 4;

  v8f acc[2][4] = {};
  const float* arow0 = A + (size_t)(row0 + am) * KD;
  const float* arow1 = A + (size_t)(row0 + 16 + am) * KD;

  for (int c0 = 0; c0 < KD; c0 += 32) {
    v16h af0 = a_frag_from_f32(arow0, c0, ag);
    v16h af1 = a_frag_from_f32(arow1, c0, ag);
#pragma unroll
    for (int t = 0; t < 4; ++t) {
      v16h bf = *(const v16h*)(W16 + (size_t)(col0 + t * 16 + am) * KD + c0 + 16 * ag);
      acc[0][t] = __builtin_amdgcn_wmma_f32_16x16x32_f16(false, af0, false, bf,
                                                         (short)0, acc[0][t], false, false);
      acc[1][t] = __builtin_amdgcn_wmma_f32_16x16x32_f16(false, af1, false, bf,
                                                         (short)0, acc[1][t], false, false);
    }
  }
  // D layout: vgpr r -> row m = r + 8g, lane n = col
  const int dn = lane & 15, dg = lane >> 4;
#pragma unroll
  for (int u = 0; u < 2; ++u) {
#pragma unroll
    for (int t = 0; t < 4; ++t) {
      int col = col0 + t * 16 + dn;
      int h = col >> 6, e = col & 63;
#pragma unroll
      for (int r = 0; r < 8; ++r) {
        int row = row0 + 16 * u + r + 8 * dg;
        int b = row >> 11, s = row & (KS - 1);
        _Float16 val = (_Float16)acc[u][t][r];
        if (mode == 0)
          dst[(((size_t)b * KNH + h) * KS + s) * KHS + e] = val;
        else
          dst[(((size_t)b * KNH + h) * KHS + e) * KS + s] = val;
      }
    }
  }
}

// =====================================================================
// Flash attention: one wave handles 16 query rows of one (b,h).
// q,k: [B,NH,S,HS] f16 ; vT: [B,NH,HS,S] f16 ; out: [B,S,D] f16
// =====================================================================
__global__ __launch_bounds__(32)
void attn_kernel(const _Float16* __restrict__ q16, const _Float16* __restrict__ k16,
                 const _Float16* __restrict__ vT16, const int* __restrict__ pad,
                 const int* __restrict__ tmask, _Float16* __restrict__ attn_out) {
  __shared__ __align__(32) _Float16 pshare[16 * 32];
  const int lane = threadIdx.x & 31;
  int bid = blockIdx.x;
  const int qt = bid & 127; bid >>= 7;
  const int h  = bid & 15;  bid >>= 4;
  const int b  = bid;
  const int q0 = qt * 16;

  const size_t head = ((size_t)(b * KNH + h)) * KS * KHS;
  const _Float16* qh = q16 + head;
  const _Float16* kh = k16 + head;
  const _Float16* vh = vT16 + head;  // [HS][S]

  const int am = lane & 15, ag = lane >> 4;   // A-layout roles
  const int nn = lane & 15, ng = lane >> 4;   // B / C-D layout roles

  // Q fragments: two 16x32 chunks over HS=64
  v16h aq[2];
  {
    const _Float16* qrow = qh + (size_t)(q0 + am) * KHS;
    aq[0] = a_frag_from_f16(qrow, 0, ag);
    aq[1] = a_frag_from_f16(qrow, 32, ag);
  }

  v8f acc[4] = {};
  float mrow[8], lrow[8];
#pragma unroll
  for (int r = 0; r < 8; ++r) { mrow[r] = -3.0e38f; lrow[r] = 0.0f; }

  const float scale = 0.125f;         // 64^-0.5
  const int jend = q0 + 16;           // causal: key j <= q0+15

  for (int jt = 0; jt < jend; jt += 32) {
    // prefetch next K tile row for this lane (speculative)
    if (jt + 32 < jend)
      __builtin_prefetch(kh + (size_t)(jt + 32 + nn) * KHS, 0, 0);

    // ---- scores: att[16 x 32] = Q @ K^T ----
    v8f s[2] = {};
#pragma unroll
    for (int tc = 0; tc < 2; ++tc) {
      const _Float16* krow = kh + (size_t)(jt + tc * 16 + nn) * KHS;
#pragma unroll
      for (int c = 0; c < 2; ++c) {
        v16h bk = *(const v16h*)(krow + 32 * c + 16 * ng);
        s[tc] = __builtin_amdgcn_wmma_f32_16x16x32_f16(false, aq[c], false, bk,
                                                       (short)0, s[tc], false, false);
      }
    }

    // ---- mask + online softmax (rows m = r + 8*ng live across lanes n=0..15) ----
    float pv[2][8];
#pragma unroll
    for (int r = 0; r < 8; ++r) {
      int row = q0 + r + 8 * ng;
      int j0 = jt + nn, j1 = jt + 16 + nn;
      bool ok0 = (j0 <= row) && (pad[b * KS + j0] != 0) && (tmask[(size_t)row * KS + j0] != 0);
      bool ok1 = (j1 <= row) && (pad[b * KS + j1] != 0) && (tmask[(size_t)row * KS + j1] != 0);
      float v0 = ok0 ? s[0][r] * scale : NEG_BIG;
      float v1 = ok1 ? s[1][r] * scale : NEG_BIG;
      float mx = fmaxf(v0, v1);
#pragma unroll
      for (int o = 1; o < 16; o <<= 1) mx = fmaxf(mx, __shfl_xor(mx, o, 32));
      float mnew = fmaxf(mrow[r], mx);
      float p0 = __expf(v0 - mnew);
      float p1 = __expf(v1 - mnew);
      pv[0][r] = p0; pv[1][r] = p1;
      float rs = p0 + p1;
#pragma unroll
      for (int o = 1; o < 16; o <<= 1) rs += __shfl_xor(rs, o, 32);
      float alpha = __expf(mrow[r] - mnew);
      lrow[r] = lrow[r] * alpha + rs;
      mrow[r] = mnew;
#pragma unroll
      for (int t = 0; t < 4; ++t) acc[t][r] *= alpha;
    }

    // ---- D-layout -> A-layout for P via LDS ----
#pragma unroll
    for (int tc = 0; tc < 2; ++tc)
#pragma unroll
      for (int r = 0; r < 8; ++r)
        pshare[(r + 8 * ng) * 32 + tc * 16 + nn] = (_Float16)pv[tc][r];
    __syncthreads();
    v16h ap = a_frag_from_f16(pshare + am * 32, 0, ag);
    __syncthreads();

    // ---- acc += P @ V  (B frag: lane e-column contiguous thanks to V^T) ----
#pragma unroll
    for (int t = 0; t < 4; ++t) {
      v16h bv = *(const v16h*)(vh + (size_t)(t * 16 + nn) * KS + jt + 16 * ng);
      acc[t] = __builtin_amdgcn_wmma_f32_16x16x32_f16(false, ap, false, bv,
                                                      (short)0, acc[t], false, false);
    }
  }

  // finalize + merge heads into [B,S,D] f16
#pragma unroll
  for (int t = 0; t < 4; ++t) {
    int e = t * 16 + nn;
#pragma unroll
    for (int r = 0; r < 8; ++r) {
      int row = q0 + r + 8 * ng;
      float o = acc[t][r] / lrow[r];
      attn_out[((size_t)b * KS + row) * KD + h * KHS + e] = (_Float16)o;
    }
  }
}

// =====================================================================
// Output projection: out(f32) = A16 @ Wp16^T + bp, 32x64 tile per wave
// =====================================================================
__global__ __launch_bounds__(128)
void outproj_kernel(const _Float16* __restrict__ A16, const _Float16* __restrict__ W16,
                    const float* __restrict__ bias, float* __restrict__ out) {
  const int lane = threadIdx.x & 31;
  const int wid  = threadIdx.x >> 5;
  const int row0 = blockIdx.x * 32;
  const int col0 = (blockIdx.y * 4 + wid) * 64;
  const int am = lane & 15, ag = lane >> 4;

  v8f acc[2][4] = {};
  const _Float16* arow0 = A16 + (size_t)(row0 + am) * KD;
  const _Float16* arow1 = A16 + (size_t)(row0 + 16 + am) * KD;

  for (int c0 = 0; c0 < KD; c0 += 32) {
    v16h af0 = a_frag_from_f16(arow0, c0, ag);
    v16h af1 = a_frag_from_f16(arow1, c0, ag);
#pragma unroll
    for (int t = 0; t < 4; ++t) {
      v16h bf = *(const v16h*)(W16 + (size_t)(col0 + t * 16 + am) * KD + c0 + 16 * ag);
      acc[0][t] = __builtin_amdgcn_wmma_f32_16x16x32_f16(false, af0, false, bf,
                                                         (short)0, acc[0][t], false, false);
      acc[1][t] = __builtin_amdgcn_wmma_f32_16x16x32_f16(false, af1, false, bf,
                                                         (short)0, acc[1][t], false, false);
    }
  }
  const int dn = lane & 15, dg = lane >> 4;
#pragma unroll
  for (int u = 0; u < 2; ++u) {
#pragma unroll
    for (int t = 0; t < 4; ++t) {
      int col = col0 + t * 16 + dn;
      float bb = bias[col];
#pragma unroll
      for (int r = 0; r < 8; ++r) {
        int row = row0 + 16 * u + r + 8 * dg;
        out[(size_t)row * KD + col] = acc[u][t][r] + bb;
      }
    }
  }
}

extern "C" void kernel_launch(void* const* d_in, const int* in_sizes, int n_in,
                              void* d_out, int out_size, void* d_ws, size_t ws_size,
                              hipStream_t stream) {
  const float* x     = (const float*)d_in[0];
  const float* y     = (const float*)d_in[1];
  const int*   pad   = (const int*)d_in[2];
  const int*   tmask = (const int*)d_in[3];
  const float* Wq    = (const float*)d_in[4];
  const float* Wk    = (const float*)d_in[5];
  const float* Wv    = (const float*)d_in[6];
  const float* Wp    = (const float*)d_in[7];
  const float* bp    = (const float*)d_in[8];
  float* out = (float*)d_out;

  _Float16* ws = (_Float16*)d_ws;
  const size_t actBuf = (size_t)KB * KS * KD;     // 8M f16 = 16 MB
  const size_t wBuf   = (size_t)KD * KD;          // 1M f16 = 2 MB
  _Float16* q16    = ws;
  _Float16* k16    = ws + actBuf;
  _Float16* vT16   = ws + 2 * actBuf;
  _Float16* attn16 = ws + 3 * actBuf;
  _Float16* wq16   = ws + 4 * actBuf;
  _Float16* wk16   = wq16 + wBuf;
  _Float16* wv16   = wk16 + wBuf;
  _Float16* wp16   = wv16 + wBuf;                 // 72 MB total

  // one-time weight conversions
  const int wN = KD * KD;
  const int cvtBlocks = wN / (256 * 8);           // 512
  cvt_kernel<<<cvtBlocks, 256, 0, stream>>>(Wq, wq16, wN);
  cvt_kernel<<<cvtBlocks, 256, 0, stream>>>(Wk, wk16, wN);
  cvt_kernel<<<cvtBlocks, 256, 0, stream>>>(Wv, wv16, wN);
  cvt_kernel<<<cvtBlocks, 256, 0, stream>>>(Wp, wp16, wN);

  dim3 gemm_grid(KB * KS / 32, 4);                // (256,4), 4 waves/block
  proj_kernel<<<gemm_grid, 128, 0, stream>>>(y, wq16, q16, 0);
  proj_kernel<<<gemm_grid, 128, 0, stream>>>(x, wk16, k16, 0);
  proj_kernel<<<gemm_grid, 128, 0, stream>>>(x, wv16, vT16, 1);

  attn_kernel<<<KB * KNH * (KS / 16), 32, 0, stream>>>(q16, k16, vT16, pad, tmask, attn16);

  outproj_kernel<<<gemm_grid, 128, 0, stream>>>(attn16, wp16, bp, out);
}
